// OperationCell_9096740733473
// MI455X (gfx1250) — matrix operand order repaired and checked by reading, not compiled, not run on hardware
//
#include <hip/hip_runtime.h>
#include <hip/hip_bf16.h>
#include <stdint.h>

// ---------------------------------------------------------------------------
// Problem constants (from reference)
// ---------------------------------------------------------------------------
#define BATCH 4096
#define INDIM 1024
#define HID   1024
#define NAGG  4
#define NOPS  29

typedef __attribute__((ext_vector_type(16))) __bf16       v16bf;
typedef __attribute__((ext_vector_type(8)))  float        v8f;
typedef __attribute__((ext_vector_type(4)))  unsigned int u32x4;
typedef __attribute__((ext_vector_type(8)))  int          i32x8;
typedef __attribute__((ext_vector_type(4)))  int          i32x4;

// ---------------------------------------------------------------------------
// f32 -> bf16 conversion (strided dest so x|h pack into one (B,2048) matrix)
// ---------------------------------------------------------------------------
__global__ void cvt_f32_bf16(const float* __restrict__ src,
                             unsigned short* __restrict__ dst,
                             long long total4, int cols, int dstStride, int dstOff)
{
    long long i = (long long)blockIdx.x * blockDim.x + threadIdx.x;
    if (i >= total4) return;
    long long e = i * 4;
    int row = (int)(e / cols);
    int col = (int)(e % cols);
    const float4 f = *(const float4*)(src + e);
    union { __bf16 b; unsigned short u; } cv;
    unsigned short s0, s1, s2, s3;
    cv.b = (__bf16)f.x; s0 = cv.u;
    cv.b = (__bf16)f.y; s1 = cv.u;
    cv.b = (__bf16)f.z; s2 = cv.u;
    cv.b = (__bf16)f.w; s3 = cv.u;
    uint2 packed;
    packed.x = (unsigned)s0 | ((unsigned)s1 << 16);
    packed.y = (unsigned)s2 | ((unsigned)s3 << 16);
    *(uint2*)(dst + (size_t)row * dstStride + dstOff + col) = packed;
}

// ---------------------------------------------------------------------------
// TDM: 2-D tile (rows x 32 bf16) global -> LDS, with LDS padding of
// 4 dwords per 16 dwords (row stride 64B -> 80B, bank-conflict-free b128 reads)
// Descriptor per CDNA5 ISA; 6-arg builtin on this
// toolchain: (g0, g1, g2, g3, g_extra, cpol).
// ---------------------------------------------------------------------------
__device__ __forceinline__ void tdm_load_tile_rx32_bf16(unsigned lds_byte,
                                                        const void* gaddr,
                                                        unsigned lda_elems,
                                                        unsigned rows)
{
    unsigned long long ga = (unsigned long long)gaddr;
    u32x4 g0;
    g0[0] = 1u;                                   // count=1, user desc, no gather
    g0[1] = lds_byte;                             // lds_addr (bytes)
    g0[2] = (unsigned)ga;                         // global_addr[31:0]
    g0[3] = (unsigned)((ga >> 32) & 0x01FFFFFFull) | (2u << 30); // addr[56:32], type=2
    i32x8 g1;
    g1[0] = (1 << 16)        // data_size = 1 (2 bytes)
          | (1 << 20)        // pad_enable
          | (3 << 22)        // pad_interval: 16 dwords (one 64B row)
          | (3 << 25);       // pad_amount: 4 dwords (16B)  -> row stride 80B
    g1[1] = (int)(32u << 16);     // tensor_dim0 = 32 (low 16 bits @ [63:48])
    g1[2] = (int)(rows << 16);    // tensor_dim0 hi = 0 | tensor_dim1 = rows (low16)
    g1[3] = (int)(32u << 16);     // tensor_dim1 hi = 0 | tile_dim0 = 32
    g1[4] = (int)rows;            // tile_dim1 = rows, tile_dim2 = 0
    g1[5] = (int)lda_elems;       // tensor_dim0_stride low32 (elements)
    g1[6] = 0;                    // stride0 hi16 | stride1 low16
    g1[7] = 0;                    // stride1 hi32
    i32x4 z4 = (i32x4)0;
    i32x8 z8 = (i32x8)0;
    __builtin_amdgcn_tensor_load_to_lds(g0, g1, z4, z4, z8, 0);
}

__device__ __forceinline__ v16bf ld_frag(const unsigned char* p0,
                                         const unsigned char* p1)
{
    union { v16bf v; uint4 q[2]; } u;
    u.q[0] = *(const uint4*)p0;    // -> ds_load_b128
    u.q[1] = *(const uint4*)p1;
    return u.v;
}

// ---------------------------------------------------------------------------
// NT bf16 GEMM: C[M,N] (f32) = A[M,K] (bf16, row-major, stride lda)
//                            x B[N,K]^T (bf16, row-major, stride ldb)
// Block tile 128x256, 8 waves (2x4), wave tile 64x64 (4x4 WMMA accumulators:
// 16 v_wmma per 16 ds_load_b128 -> 1:1 LDS/WMMA issue ratio).
// Double-buffered LDS staging via TDM; 80-byte padded LDS rows.
// ---------------------------------------------------------------------------
#define BM 128
#define BN 256
#define BK 32
#define LDS_ROW 80

__global__ __launch_bounds__(256) void gemm_bf16_wmma(
    const __bf16* __restrict__ A, int lda,
    const __bf16* __restrict__ B, int ldb,
    float* __restrict__ C, int ldc, int K)
{
    __shared__ __align__(16) unsigned char smA[2][BM * LDS_ROW];   // 2 x 10 KB
    __shared__ __align__(16) unsigned char smB[2][BN * LDS_ROW];   // 2 x 20 KB

    const int tid  = threadIdx.x;
    const int wave = tid >> 5;
    const int lane = tid & 31;
    const int half = lane >> 4;   // K-half selector within WMMA fragments
    const int r    = lane & 15;   // row (A) / column (B) / N (C) index
    const int wm   = wave >> 2;   // 0..1  -> 64-row slab
    const int wn   = wave & 3;    // 0..3  -> 64-col slab

    const int m0 = blockIdx.x * BM;
    const int n0 = blockIdx.y * BN;

    v8f acc[4][4];
#pragma unroll
    for (int mi = 0; mi < 4; ++mi)
#pragma unroll
        for (int ni = 0; ni < 4; ++ni)
#pragma unroll
            for (int q = 0; q < 8; ++q) acc[mi][ni][q] = 0.0f;

    const int nk = K / BK;
    unsigned ldsA[2] = { (unsigned)(unsigned long long)&smA[0][0],
                         (unsigned)(unsigned long long)&smA[1][0] };
    unsigned ldsB[2] = { (unsigned)(unsigned long long)&smB[0][0],
                         (unsigned)(unsigned long long)&smB[1][0] };

    if (wave == 0) {
        tdm_load_tile_rx32_bf16(ldsA[0], A + (size_t)m0 * lda, (unsigned)lda, BM);
        tdm_load_tile_rx32_bf16(ldsB[0], B + (size_t)n0 * ldb, (unsigned)ldb, BN);
    }

    for (int kt = 0; kt < nk; ++kt) {
        __syncthreads();                       // everyone done reading buf[kt+1 & 1]
        if (wave == 0) {
            if (kt + 1 < nk) {
                const int k1 = (kt + 1) * BK;
                tdm_load_tile_rx32_bf16(ldsA[(kt + 1) & 1],
                                        A + (size_t)m0 * lda + k1, (unsigned)lda, BM);
                tdm_load_tile_rx32_bf16(ldsB[(kt + 1) & 1],
                                        B + (size_t)n0 * ldb + k1, (unsigned)ldb, BN);
                __builtin_amdgcn_s_wait_tensorcnt(2);  // current buffer's 2 DMAs done
            } else {
                __builtin_amdgcn_s_wait_tensorcnt(0);
            }
        }
        __syncthreads();                       // publish LDS tile to all waves

        const unsigned char* bufA = smA[kt & 1];
        const unsigned char* bufB = smB[kt & 1];

        // B fragments: 32x16 bf16; lane r = column, halves hold K 0..15 / 16..31
        v16bf bfrag[4];
#pragma unroll
        for (int ni = 0; ni < 4; ++ni) {
            const unsigned char* base =
                bufB + (size_t)(wn * 64 + ni * 16 + r) * LDS_ROW + half * 32;
            bfrag[ni] = ld_frag(base, base + 16);
        }
        // A fragments: 16x32 bf16; lane r = row, halves hold K 0..15 / 16..31
#pragma unroll
        for (int mi = 0; mi < 4; ++mi) {
            const unsigned char* base =
                bufA + (size_t)(wm * 64 + mi * 16 + r) * LDS_ROW + half * 32;
            v16bf afrag = ld_frag(base, base + 16);
#pragma unroll
            for (int ni = 0; ni < 4; ++ni) {
                acc[mi][ni] = __builtin_amdgcn_wmma_f32_16x16x32_bf16(
                    false, afrag, false, bfrag[ni],
                    (short)0, acc[mi][ni], false, false);
            }
        }
    }

    // C/D layout: lane(0-15) n, lane(16-31) n with m+8; VGPR v holds row v (+8)
#pragma unroll
    for (int mi = 0; mi < 4; ++mi) {
#pragma unroll
        for (int ni = 0; ni < 4; ++ni) {
            const int mbase = m0 + wm * 64 + mi * 16 + half * 8;
            const int nb    = n0 + wn * 64 + ni * 16 + r;
#pragma unroll
            for (int v = 0; v < 8; ++v)
                C[(size_t)(mbase + v) * ldc + nb] = acc[mi][ni][v];
        }
    }
}

// ---------------------------------------------------------------------------
// Gumbel argmax + selected-op evaluation (forward of straight-through == one-hot)
// ---------------------------------------------------------------------------
__device__ __forceinline__ float sgnf(float v)
{ return v > 0.f ? 1.f : (v < 0.f ? -1.f : 0.f); }

__device__ __forceinline__ float safe_div(float a, float b)
{ return a / (b + 1e-6f * sgnf(b + 1e-9f)); }

__global__ __launch_bounds__(256) void select_ops(
    const float* __restrict__ xp_, const float* __restrict__ hp_,
    const float* __restrict__ agg_, const float* __restrict__ opw,
    const float* __restrict__ gumbel, const float* __restrict__ cst,
    const float* __restrict__ ew, float* __restrict__ out)
{
    __shared__ float z[256 * NOPS];
    const int flat0 = blockIdx.x * 256;            // covers one b, 256 consecutive j
    const int b  = flat0 >> 10;
    const int j0 = flat0 & (HID - 1);
    const float* gsrc = gumbel + (size_t)flat0 * NOPS;
    const float* wsrc = opw + (size_t)j0 * NOPS;
    for (int i = threadIdx.x; i < 256 * NOPS; i += 256)
        z[i] = wsrc[i] + gsrc[i];                  // coalesced gumbel stream (dominant)
    __syncthreads();

    const int tid = threadIdx.x;
    const int j = j0 + tid;
    const float* zz = z + tid * NOPS;
    int best = 0; float bv = zz[0];
#pragma unroll
    for (int o = 1; o < NOPS; ++o) { float v = zz[o]; if (v > bv) { bv = v; best = o; } }

    const float xv = xp_[(size_t)b * HID + j];
    const float hv = hp_[(size_t)b * HID + j];
    const float c  = cst[j];
    float val;
    if (best < 22) {
        switch (best) {
            case 0:  val = xv; break;
            case 1:  val = xv + c; break;
            case 2:  val = xv * c; break;
            case 3:  val = xv * xv; break;
            case 4:  val = xv * xv * xv; break;
            case 5:  val = sqrtf(fabsf(xv) + 1e-6f); break;
            case 6:  val = fabsf(xv); break;
            case 7:  val = -xv; break;
            case 8:  val = sinf(xv); break;
            case 9:  val = cosf(xv); break;
            case 10: val = tanhf(xv); break;
            case 11: val = 1.f / (1.f + expf(-xv)); break;
            case 12: val = expf(fminf(fmaxf(xv, -10.f), 10.f)); break;
            case 13: val = logf(fabsf(xv) + 1e-6f); break;
            case 14: val = safe_div(1.f, xv); break;
            case 15: val = xv + hv; break;
            case 16: val = xv - hv; break;
            case 17: val = xv * hv; break;
            case 18: val = safe_div(xv, hv); break;
            case 19: val = fmaxf(xv, hv); break;
            case 20: val = fminf(xv, hv); break;
            default: val = sgnf(xv) *
                           powf(fabsf(xv) + 1e-6f, fminf(fmaxf(hv, -3.f), 3.f));
        }
    } else {
        const float* ar = agg_ + (size_t)b * (HID * NAGG) + (size_t)j * NAGG;
        const float* er = ew + (size_t)j * NAGG;
        float w[NAGG], s = 0.f;
#pragma unroll
        for (int a = 0; a < NAGG; ++a) { w[a] = ar[a] * er[a]; s += w[a]; }
        switch (best) {
            case 22: val = s; break;
            case 23: val = s * 0.25f; break;
            case 24: {
                float p = 1.f;
#pragma unroll
                for (int a = 0; a < NAGG; ++a)
                    p *= fminf(fmaxf(w[a], -10.f), 10.f);
                val = p; break;
            }
            case 25: val = fmaxf(fmaxf(w[0], w[1]), fmaxf(w[2], w[3])); break;
            case 26: val = fminf(fminf(w[0], w[1]), fminf(w[2], w[3])); break;
            case 27: {
                float m = s * 0.25f, var = 0.f;
#pragma unroll
                for (int a = 0; a < NAGG; ++a) { float d = w[a] - m; var += d * d; }
                val = sqrtf(var * (1.f / 3.f)) + 1e-6f; break;
            }
            default: {
                float q = 0.f;
#pragma unroll
                for (int a = 0; a < NAGG; ++a) q += w[a] * w[a];
                val = sqrtf(q);
            }
        }
    }
    out[(size_t)flat0 + tid] = fminf(fmaxf(val, -100.f), 100.f);
}

// ---------------------------------------------------------------------------
// Entropy of softmax(op_weights/TAU) averaged over HID (TAU == 1)
// ---------------------------------------------------------------------------
__global__ void entropy_k(const float* __restrict__ opw, float* __restrict__ outp)
{
    __shared__ float red[256];
    float acc = 0.f;
    for (int j = threadIdx.x; j < HID; j += 256) {
        const float* row = opw + (size_t)j * NOPS;
        float m = row[0];
        for (int o = 1; o < NOPS; ++o) m = fmaxf(m, row[o]);
        float ex[NOPS], den = 0.f;
        for (int o = 0; o < NOPS; ++o) { ex[o] = expf(row[o] - m); den += ex[o]; }
        float inv = 1.f / den, ent = 0.f;
        for (int o = 0; o < NOPS; ++o) { float p = ex[o] * inv; ent -= p * logf(p + 1e-10f); }
        acc += ent;
    }
    red[threadIdx.x] = acc;
    __syncthreads();
    for (int s = 128; s > 0; s >>= 1) {
        if (threadIdx.x < s) red[threadIdx.x] += red[threadIdx.x + s];
        __syncthreads();
    }
    if (threadIdx.x == 0) outp[0] = red[0] / (float)HID;
}

// ---------------------------------------------------------------------------
// Launch
// ---------------------------------------------------------------------------
extern "C" void kernel_launch(void* const* d_in, const int* in_sizes, int n_in,
                              void* d_out, int out_size, void* d_ws, size_t ws_size,
                              hipStream_t stream)
{
    const float* x    = (const float*)d_in[0];
    const float* h    = (const float*)d_in[1];
    const float* opw  = (const float*)d_in[2];
    const float* cst  = (const float*)d_in[3];
    const float* Win  = (const float*)d_in[4];
    const float* Wh   = (const float*)d_in[5];
    const float* ew   = (const float*)d_in[6];
    const float* Wagg = (const float*)d_in[7];
    const float* gum  = (const float*)d_in[8];
    float* out = (float*)d_out;

    // Workspace layout (~132 MB): bf16 operands + f32 GEMM outputs
    unsigned char* ws = (unsigned char*)d_ws;
    unsigned short* combined = (unsigned short*)(ws);               // (B, 2048) bf16
    unsigned short* winb  = (unsigned short*)(ws + 16777216);       // (1024,1024) bf16
    unsigned short* whb   = (unsigned short*)(ws + 18874368);       // (1024,1024) bf16
    unsigned short* waggb = (unsigned short*)(ws + 20971520);       // (4096,2048) bf16
    float* xp  = (float*)(ws + 37748736);                           // (B, HID) f32
    float* hp  = (float*)(ws + 54525952);                           // (B, HID) f32
    float* agg = (float*)(ws + 71303168);                           // (B, 4096) f32

    auto blocks4 = [](long long n) { return (unsigned)((n / 4 + 255) / 256); };
    cvt_f32_bf16<<<blocks4(4096LL * 1024), 256, 0, stream>>>(x, combined, 4096LL * 1024 / 4, 1024, 2048, 0);
    cvt_f32_bf16<<<blocks4(4096LL * 1024), 256, 0, stream>>>(h, combined, 4096LL * 1024 / 4, 1024, 2048, 1024);
    cvt_f32_bf16<<<blocks4(1024LL * 1024), 256, 0, stream>>>(Win, winb, 1024LL * 1024 / 4, 1024, 1024, 0);
    cvt_f32_bf16<<<blocks4(1024LL * 1024), 256, 0, stream>>>(Wh, whb, 1024LL * 1024 / 4, 1024, 1024, 0);
    cvt_f32_bf16<<<blocks4(4096LL * 2048), 256, 0, stream>>>(Wagg, waggb, 4096LL * 2048 / 4, 2048, 2048, 0);

    // xp = x @ W_in^T   (A = combined cols [0,1024), lda=2048)
    gemm_bf16_wmma<<<dim3(BATCH / BM, HID / BN), 256, 0, stream>>>(
        (const __bf16*)combined, 2048, (const __bf16*)winb, 1024, xp, HID, 1024);
    // hp = h @ W_h^T    (A = combined cols [1024,2048))
    gemm_bf16_wmma<<<dim3(BATCH / BM, HID / BN), 256, 0, stream>>>(
        (const __bf16*)combined + 1024, 2048, (const __bf16*)whb, 1024, hp, HID, 1024);
    // agg = [x|h] @ W_agg^T
    gemm_bf16_wmma<<<dim3(BATCH / BM, (HID * NAGG) / BN), 256, 0, stream>>>(
        (const __bf16*)combined, 2048, (const __bf16*)waggb, 2048, agg, HID * NAGG, 2048);

    select_ops<<<(BATCH * HID) / 256, 256, 0, stream>>>(xp, hp, agg, opw, gum, cst, ew, out);
    entropy_k<<<1, 256, 0, stream>>>(opw, out + (size_t)BATCH * HID);
}
